// LundVicReg_11974368821701
// MI455X (gfx1250) — compile-verified, matrix operand order checked by reading, not run on hardware
//
#include <hip/hip_runtime.h>
#include <hip/hip_bf16.h>

// LundNet-style GNN for MI455X (gfx1250, wave32).
// All large edge-GEMMs ([E,2c]x[2c,c']) run on v_wmma_f32_16x16x32_bf16 with
// fp32 accumulation. Feature tables (<=25.6MB) are L2-resident, so gathers and
// scatter-atomics stay on-chip; BatchNorm uses a recompute two-pass scheme
// (stats pass + apply/scatter pass) to avoid spilling the [E,c] edge tensor to
// HBM. K is a template parameter so A-fragments are register-resident across
// the whole output-channel loop (2 ds_load_b128 per WMMA instead of 4).

typedef __bf16 bf16;
typedef __attribute__((ext_vector_type(16))) __bf16 v16bf;
typedef __attribute__((ext_vector_type(8)))  float  v8f;

#define BN_EPS 1e-5f

// ---------------------------------------------------------------- utilities
__global__ void fill_zero(float* p, size_t n) {
    size_t i = (size_t)blockIdx.x * blockDim.x + threadIdx.x;
    if (i < n) p[i] = 0.f;
}

__global__ void scatter_ones(const int* __restrict__ idx, float* out, int n) {
    int i = blockIdx.x * blockDim.x + threadIdx.x;
    if (i < n) atomicAdd(&out[idx[i]], 1.0f);
}

// per-column sum / sum-of-squares (c <= 128), grid-stride + LDS reduce
__global__ void colstats(const float* __restrict__ x, int rows, int c, float* stats) {
    __shared__ float ss[128], sq[128];
    int tid = threadIdx.x;
    if (tid < c) { ss[tid] = 0.f; sq[tid] = 0.f; }
    __syncthreads();
    size_t n = (size_t)rows * c;
    for (size_t i = (size_t)blockIdx.x * blockDim.x + tid; i < n;
         i += (size_t)gridDim.x * blockDim.x) {
        float v = x[i];
        int col = (int)(i % c);
        atomicAdd(&ss[col], v);
        atomicAdd(&sq[col], v * v);
    }
    __syncthreads();
    if (tid < c) { atomicAdd(&stats[tid], ss[tid]); atomicAdd(&stats[c + tid], sq[tid]); }
}

// stats -> folded BN affine: scale = g*rsqrt(var+eps), shift = b - mean*scale
__global__ void finalize_bn(const float* __restrict__ stats, float cnt,
                            const float* __restrict__ g, const float* __restrict__ b,
                            float* scale, float* shift, int c) {
    int j = threadIdx.x;
    if (j < c) {
        float m   = stats[j] / cnt;
        float var = stats[c + j] / cnt - m * m;
        float a   = g[j] * rsqrtf(var + BN_EPS);
        scale[j] = a;
        shift[j] = b[j] - m * a;
    }
}

__global__ void affine_cols(const float* __restrict__ x, const float* __restrict__ sc,
                            const float* __restrict__ sh, float* out, int rows, int c) {
    int i = blockIdx.x * blockDim.x + threadIdx.x;
    if (i < rows * c) { int col = i % c; out[i] = x[i] * sc[col] + sh[col]; }
}

// fts = relu(acc / max(deg,1))
__global__ void node_finalize(const float* __restrict__ acc, const float* __restrict__ deg,
                              float* out, int rows, int c, int cshift) {
    int i = blockIdx.x * blockDim.x + threadIdx.x;
    if (i < rows * c) {
        int r = i >> cshift;
        float v = acc[i] / fmaxf(deg[r], 1.f);
        out[i] = fmaxf(v, 0.f);
    }
}

__global__ void residual_k(const float* __restrict__ sc, const float* __restrict__ x,
                           float* out, int n) {
    int i = blockIdx.x * blockDim.x + threadIdx.x;
    if (i < n) out[i] = fmaxf(sc[i] + x[i], 0.f);
}

__global__ void scatter_pool(const float* __restrict__ fts, const int* __restrict__ ng,
                             float* pool, int N, int c, int cshift) {
    int i = blockIdx.x * blockDim.x + threadIdx.x;
    if (i < N * c) {
        int r = i >> cshift;
        atomicAdd(&pool[(size_t)ng[r] * c + (i & (c - 1))], fts[i]);
    }
}

__global__ void pool_div(float* pool, const float* __restrict__ npg, int total, int cshift) {
    int i = blockIdx.x * blockDim.x + threadIdx.x;
    if (i < total) pool[i] /= fmaxf(npg[i >> cshift], 1.f);
}

template <bool RELU>
__global__ void linear_k(const float* __restrict__ x, const float* __restrict__ w,
                         const float* __restrict__ b, float* out, int rows, int K, int J) {
    int idx = blockIdx.x * blockDim.x + threadIdx.x;
    if (idx >= rows * J) return;
    int g = idx / J, j = idx % J;
    const float* xr = x + (size_t)g * K;
    const float* wr = w + (size_t)j * K;
    float v = b[j];
    for (int k = 0; k < K; ++k) v += xr[k] * wr[k];
    out[idx] = RELU ? fmaxf(v, 0.f) : v;
}

// --------------------------------------------------- weight prep (fp32->bf16)
// Wt[j][k] (pitch KP): k<cin -> theta, k<2cin -> phi-theta, else 0.  bias = tb+pb
__global__ void prep_edge_w(const float* __restrict__ tw, const float* __restrict__ pw,
                            const float* __restrict__ tb, const float* __restrict__ pb,
                            int cin, int cout, int KP, bf16* Wt, float* bias) {
    int idx = blockIdx.x * blockDim.x + threadIdx.x;
    if (idx < cout * KP) {
        int j = idx / KP, k = idx % KP;
        float v = 0.f;
        if (k < cin)          v = tw[j * cin + k];
        else if (k < 2 * cin) v = pw[j * cin + (k - cin)] - tw[j * cin + (k - cin)];
        Wt[idx] = (bf16)v;
    }
    if (idx < cout) bias[idx] = tb[idx] + pb[idx];
}

__global__ void prep_node_w(const float* __restrict__ w, int K, int cout, int KP, bf16* Wt) {
    int idx = blockIdx.x * blockDim.x + threadIdx.x;
    if (idx < cout * KP) {
        int j = idx / KP, k = idx % KP;
        Wt[idx] = (bf16)((k < K) ? w[j * K + k] : 0.f);
    }
}

// ------------------------------------------------------- WMMA edge/node GEMM
// GATHER=true : rows are edges; A-row = [fts[dst] | fts[src]]  (K = 2*cin)
// GATHER=false: rows are nodes; A-row = fts[row]               (K = cin)
// PASS 0: accumulate per-channel sum / sumsq of e (with linear bias) -> stats
// PASS 1: apply folded BN affine; GATHER -> atomic scatter to out[dst][n],
//         else direct store out[row][n].
// Block = 256 threads = 8 waves; each wave owns one 16-row M-tile and keeps
// its KT/32 A-fragments register-resident across the whole nt loop.
template <int PASS, bool GATHER, int KT>
__global__ __launch_bounds__(256) void edge_wmma(
    const float* __restrict__ fts, const int* __restrict__ src, const int* __restrict__ dst,
    const bf16* __restrict__ Wt, const float* __restrict__ bias,
    float* __restrict__ stats, const float* __restrict__ scale,
    const float* __restrict__ shift, float* __restrict__ out,
    int cout, int rows) {
    constexpr int K      = KT;
    constexpr int KP     = KT + 16;              // +16 bf16 pad: 32B, kills bank conflicts
    constexpr int NK     = KT / 32;              // WMMA K-steps
    constexpr int KSHIFT = (KT == 32) ? 5 : (KT == 64) ? 6 : (KT == 128) ? 7 : 8;
    constexpr int CIN    = KT / 2;

    extern __shared__ char smem[];
    bf16* zs = (bf16*)smem;                      // 128 * KP   gathered rows (bf16)
    bf16* ws = zs + 128 * KP;                    // cout * KP  weights (bf16)
    float* s_sum = (float*)(ws + cout * KP);     // cout
    float* s_sq  = s_sum + cout;                 // cout
    int*   s_dst = (int*)(s_sq + cout);          // 128

    const int tid  = threadIdx.x;
    const int row0 = blockIdx.x * 128;
    const int nrows = min(128, rows - row0);

    // weights -> LDS (both pitched KP: linear 16B copy)
    {
        const uint4* gw = (const uint4*)Wt;
        uint4* lw = (uint4*)ws;
        int cnt = cout * KP / 8;
        for (int i = tid; i < cnt; i += 256) lw[i] = gw[i];
    }
    // gather + fp32->bf16 convert rows into LDS
    {
        int total = nrows * K;
        for (int i = tid; i < total; i += 256) {
            int r = i >> KSHIFT, k = i & (K - 1);
            float v;
            if (GATHER) {
                int node = (k < CIN) ? dst[row0 + r] : src[row0 + r];
                v = fts[(size_t)node * CIN + (k & (CIN - 1))];
            } else {
                v = fts[(size_t)(row0 + r) * K + k];
            }
            zs[r * KP + k] = (bf16)v;
        }
        if (GATHER && tid < nrows) s_dst[tid] = dst[row0 + tid];
    }
    if (PASS == 0 && tid < cout) { s_sum[tid] = 0.f; s_sq[tid] = 0.f; }
    __syncthreads();

    const int wave = tid >> 5;
    const int lane = tid & 31;
    const int half = lane >> 4;                  // K-half select (ISA 7.12.2)
    const int l16  = lane & 15;
    const int tiles = nrows >> 4;                // nrows always multiple of 16

    if (wave < tiles) {                          // uniform per wave: EXEC all-1s for WMMA
        union Frag { v16bf v; uint4 q[2]; };
        // A fragments: load once, keep in registers for all output tiles.
        // Layout (16-bit A 16x32): lanes 0-15 hold K k0+0..7 & k0+16..23,
        // lanes 16-31 hold K k0+8..15 & k0+24..31 -> two 16B LDS loads each.
        const bf16* zrow = zs + (wave * 16 + l16) * KP;
        Frag af[NK];
#pragma unroll
        for (int kk = 0; kk < NK; ++kk) {
            af[kk].q[0] = *(const uint4*)(zrow + kk * 32 + 8 * half);
            af[kk].q[1] = *(const uint4*)(zrow + kk * 32 + 16 + 8 * half);
        }
        for (int nt = 0; nt < (cout >> 4); ++nt) {
            const bf16* wrow = ws + (nt * 16 + l16) * KP;
            v8f acc = {};
#pragma unroll
            for (int kk = 0; kk < NK; ++kk) {
                Frag b;
                b.q[0] = *(const uint4*)(wrow + kk * 32 + 8 * half);
                b.q[1] = *(const uint4*)(wrow + kk * 32 + 16 + 8 * half);
                acc = __builtin_amdgcn_wmma_f32_16x16x32_bf16(
                    false, af[kk].v, false, b.v, (short)0, acc, false, false);
            }
            // C/D layout: lane -> N = l16; element r -> M = r + 8*half
            int n = nt * 16 + l16;
            float bs = bias ? bias[n] : 0.f;
            if (PASS == 0) {
                float s = 0.f, s2 = 0.f;
#pragma unroll
                for (int r = 0; r < 8; ++r) { float v = acc[r] + bs; s += v; s2 += v * v; }
                atomicAdd(&s_sum[n], s);
                atomicAdd(&s_sq[n], s2);
            } else {
                float aa = scale[n], sh = shift[n];
#pragma unroll
                for (int r = 0; r < 8; ++r) {
                    float v = (acc[r] + bs) * aa + sh;
                    int ml = wave * 16 + r + 8 * half;
                    if (GATHER)
                        atomicAdd(&out[(size_t)s_dst[ml] * cout + n], v);
                    else
                        out[(size_t)(row0 + ml) * cout + n] = v;
                }
            }
        }
    }
    if (PASS == 0) {
        __syncthreads();
        if (tid < cout) {
            atomicAdd(&stats[tid], s_sum[tid]);
            atomicAdd(&stats[cout + tid], s_sq[tid]);
        }
    }
}

// ------------------------------------------- scalar kernels for c_in=5 layers
template <int PASS>
__global__ void conv5(const float* __restrict__ fts, const int* __restrict__ src,
                      const int* __restrict__ dst, const float* __restrict__ tw,
                      const float* __restrict__ tb, const float* __restrict__ pw,
                      const float* __restrict__ pb, float* stats,
                      const float* __restrict__ scale, const float* __restrict__ shift,
                      float* __restrict__ out, int E) {
    __shared__ float ss[32], sq[32];
    int tid = threadIdx.x;
    if (PASS == 0 && tid < 32) { ss[tid] = 0.f; sq[tid] = 0.f; }
    __syncthreads();
    int e = blockIdx.x * blockDim.x + tid;
    if (e < E) {
        int s = src[e], d = dst[e];
        float xs[5], xd[5];
#pragma unroll
        for (int k = 0; k < 5; ++k) { xs[k] = fts[s * 5 + k]; xd[k] = fts[d * 5 + k]; }
#pragma unroll
        for (int j = 0; j < 32; ++j) {
            float v = tb[j] + pb[j];
#pragma unroll
            for (int k = 0; k < 5; ++k)
                v += (xd[k] - xs[k]) * tw[j * 5 + k] + xs[k] * pw[j * 5 + k];
            if (PASS == 0) { atomicAdd(&ss[j], v); atomicAdd(&sq[j], v * v); }
            else atomicAdd(&out[(size_t)d * 32 + j], v * scale[j] + shift[j]);
        }
    }
    if (PASS == 0) {
        __syncthreads();
        if (tid < 32) { atomicAdd(&stats[tid], ss[tid]); atomicAdd(&stats[32 + tid], sq[tid]); }
    }
}

template <int PASS>
__global__ void sc5(const float* __restrict__ x, const float* __restrict__ w, float* stats,
                    const float* __restrict__ scale, const float* __restrict__ shift,
                    float* __restrict__ out, int N) {
    __shared__ float ss[32], sq[32];
    int tid = threadIdx.x;
    if (PASS == 0 && tid < 32) { ss[tid] = 0.f; sq[tid] = 0.f; }
    __syncthreads();
    int i = blockIdx.x * blockDim.x + tid;
    if (i < N) {
        float xv[5];
#pragma unroll
        for (int k = 0; k < 5; ++k) xv[k] = x[i * 5 + k];
#pragma unroll
        for (int j = 0; j < 32; ++j) {
            float v = 0.f;
#pragma unroll
            for (int k = 0; k < 5; ++k) v += xv[k] * w[j * 5 + k];
            if (PASS == 0) { atomicAdd(&ss[j], v); atomicAdd(&sq[j], v * v); }
            else out[(size_t)i * 32 + j] = v * scale[j] + shift[j];
        }
    }
    if (PASS == 0) {
        __syncthreads();
        if (tid < 32) { atomicAdd(&stats[tid], ss[tid]); atomicAdd(&stats[32 + tid], sq[tid]); }
    }
}

// =========================================================== host orchestration
extern "C" void kernel_launch(void* const* d_in, const int* in_sizes, int n_in,
                              void* d_out, int out_size, void* d_ws, size_t ws_size,
                              hipStream_t stream) {
    const float* features = (const float*)d_in[0];
    const int*   esrc     = (const int*)d_in[1];
    const int*   edst     = (const int*)d_in[2];
    const int*   ngraph   = (const int*)d_in[3];
    const float* bng      = (const float*)d_in[4];
    const float* bnb      = (const float*)d_in[5];
    const int E  = in_sizes[1];
    const int Nn = in_sizes[3];
    const int G  = out_size / 1024;

    // ---- walk the flattened param dict (blocks then head) ----
    struct Conv { const float *tw, *tb, *pw, *pb, *g, *b; int ci, co; };
    Conv convs[12];
    const float *scw[6] = {}, *scg[6] = {}, *scb[6] = {};
    bool hasSC[6];
    const int couts[6] = {32, 32, 64, 64, 128, 128};
    int cins[6];
    int p = 6, ci = 5;
    for (int b = 0; b < 6; ++b) {
        cins[b] = ci;
        int co = couts[b];
        for (int cv = 0; cv < 2; ++cv) {
            Conv& C = convs[b * 2 + cv];
            C.ci = (cv == 0) ? ci : co; C.co = co;
            C.tw = (const float*)d_in[p++]; C.tb = (const float*)d_in[p++];
            C.pw = (const float*)d_in[p++]; C.pb = (const float*)d_in[p++];
            C.g  = (const float*)d_in[p++]; C.b  = (const float*)d_in[p++];
        }
        hasSC[b] = (ci != co);
        if (hasSC[b]) {
            scw[b] = (const float*)d_in[p++];
            scg[b] = (const float*)d_in[p++];
            scb[b] = (const float*)d_in[p++];
        }
        ci = co;
    }
    const float *hw[3], *hb[3];
    for (int i = 0; i < 3; ++i) { hw[i] = (const float*)d_in[p++]; hb[i] = (const float*)d_in[p++]; }

    // ---- workspace bump allocator (~131 MB total) ----
    char* wsp = (char*)d_ws;
    auto balloc = [&](size_t bytes) -> void* {
        void* r = wsp; wsp += (bytes + 255) & ~(size_t)255; return r;
    };
    float* deg  = (float*)balloc((size_t)Nn * 4);
    float* npg  = (float*)balloc((size_t)G * 4);
    float* A    = (float*)balloc((size_t)Nn * 128 * 4);   // block input / residual out
    float* B    = (float*)balloc((size_t)Nn * 128 * 4);   // after conv0
    float* C    = (float*)balloc((size_t)Nn * 128 * 4);   // after conv1
    float* SC   = (float*)balloc((size_t)Nn * 128 * 4);   // shortcut
    float* acc  = (float*)balloc((size_t)Nn * 128 * 4);   // scatter accumulator
    float* stats  = (float*)balloc(256 * 4);
    float* scaleb = (float*)balloc(128 * 4);
    float* shiftb = (float*)balloc(128 * 4);
    float* biasb  = (float*)balloc(128 * 4);
    bf16*  Wt     = (bf16*)balloc((size_t)128 * 272 * 2);
    float* pool = (float*)balloc((size_t)G * 128 * 4);
    float* h1   = (float*)balloc((size_t)G * 256 * 4);
    float* h2   = (float*)balloc((size_t)G * 512 * 4);

    auto zero = [&](float* ptr, size_t n) {
        fill_zero<<<dim3((unsigned)((n + 255) / 256)), dim3(256), 0, stream>>>(ptr, n);
    };
    auto cshift_of = [](int c) { return (c == 32) ? 5 : (c == 64) ? 6 : 7; };
    auto smem_of = [](int K, int co) -> size_t {
        return (size_t)(128 + co) * (K + 16) * 2 + (size_t)2 * co * 4 + 128 * 4 + 64;
    };

#define EW_EDGE(PASS, KT, STATS, SCALE, SHIFT, OUTP)                                   \
    edge_wmma<PASS, true, KT><<<dim3((E + 127) / 128), dim3(256), smem, stream>>>(     \
        in, esrc, edst, Wt, biasb, STATS, SCALE, SHIFT, OUTP, co, E)
#define EW_NODE(PASS, KT, STATS, SCALE, SHIFT, OUTP)                                   \
    edge_wmma<PASS, false, KT><<<dim3((Nn + 127) / 128), dim3(256), smem, stream>>>(   \
        in, nullptr, nullptr, Wt, nullptr, STATS, SCALE, SHIFT, OUTP, co, Nn)

    // WMMA edge conv: in (fts [N,ci]) -> outfts (relu(BN-scatter)/deg)
    auto run_wmma_conv = [&](const Conv& Cv, const float* in, float* outfts) {
        int K = 2 * Cv.ci, co = Cv.co, KP = K + 16;
        prep_edge_w<<<dim3((co * KP + 255) / 256), dim3(256), 0, stream>>>(
            Cv.tw, Cv.pw, Cv.tb, Cv.pb, Cv.ci, co, KP, Wt, biasb);
        zero(stats, 2 * co);
        size_t smem = smem_of(K, co);
        switch (K) {
            case 64:  EW_EDGE(0, 64,  stats, nullptr, nullptr, nullptr); break;
            case 128: EW_EDGE(0, 128, stats, nullptr, nullptr, nullptr); break;
            default:  EW_EDGE(0, 256, stats, nullptr, nullptr, nullptr); break;
        }
        finalize_bn<<<dim3(1), dim3(128), 0, stream>>>(stats, (float)E, Cv.g, Cv.b, scaleb, shiftb, co);
        zero(acc, (size_t)Nn * co);
        switch (K) {
            case 64:  EW_EDGE(1, 64,  nullptr, scaleb, shiftb, acc); break;
            case 128: EW_EDGE(1, 128, nullptr, scaleb, shiftb, acc); break;
            default:  EW_EDGE(1, 256, nullptr, scaleb, shiftb, acc); break;
        }
        int tn = Nn * co;
        node_finalize<<<dim3((tn + 255) / 256), dim3(256), 0, stream>>>(
            acc, deg, outfts, Nn, co, cshift_of(co));
    };
    // WMMA node shortcut: SC = BN(in @ w^T)
    auto run_wmma_sc = [&](const float* w, const float* g, const float* b, int cin, int co,
                           const float* in, float* outsc) {
        int K = cin, KP = K + 16;
        prep_node_w<<<dim3((co * KP + 255) / 256), dim3(256), 0, stream>>>(w, K, co, KP, Wt);
        zero(stats, 2 * co);
        size_t smem = smem_of(K, co);
        switch (K) {
            case 32: EW_NODE(0, 32, stats, nullptr, nullptr, nullptr); break;
            default: EW_NODE(0, 64, stats, nullptr, nullptr, nullptr); break;
        }
        finalize_bn<<<dim3(1), dim3(128), 0, stream>>>(stats, (float)Nn, g, b, scaleb, shiftb, co);
        switch (K) {
            case 32: EW_NODE(1, 32, nullptr, scaleb, shiftb, outsc); break;
            default: EW_NODE(1, 64, nullptr, scaleb, shiftb, outsc); break;
        }
    };

    // ---- degrees / nodes-per-graph ----
    zero(deg, Nn);
    scatter_ones<<<dim3((E + 255) / 256), dim3(256), 0, stream>>>(edst, deg, E);
    zero(npg, G);
    scatter_ones<<<dim3((Nn + 255) / 256), dim3(256), 0, stream>>>(ngraph, npg, Nn);

    // ---- input BN (5 channels) -> A [N,5] ----
    zero(stats, 16);
    colstats<<<dim3(512), dim3(256), 0, stream>>>(features, Nn, 5, stats);
    finalize_bn<<<dim3(1), dim3(128), 0, stream>>>(stats, (float)Nn, bng, bnb, scaleb, shiftb, 5);
    affine_cols<<<dim3((Nn * 5 + 255) / 256), dim3(256), 0, stream>>>(
        features, scaleb, shiftb, A, Nn, 5);

    // ---- block 0 (5->32, scalar first conv + scalar shortcut) ----
    {
        zero(stats, 64);
        sc5<0><<<dim3((Nn + 255) / 256), dim3(256), 0, stream>>>(A, scw[0], stats, nullptr, nullptr, nullptr, Nn);
        finalize_bn<<<dim3(1), dim3(128), 0, stream>>>(stats, (float)Nn, scg[0], scb[0], scaleb, shiftb, 32);
        sc5<1><<<dim3((Nn + 255) / 256), dim3(256), 0, stream>>>(A, scw[0], nullptr, scaleb, shiftb, SC, Nn);

        const Conv& C0 = convs[0];
        zero(stats, 64);
        conv5<0><<<dim3((E + 255) / 256), dim3(256), 0, stream>>>(
            A, esrc, edst, C0.tw, C0.tb, C0.pw, C0.pb, stats, nullptr, nullptr, nullptr, E);
        finalize_bn<<<dim3(1), dim3(128), 0, stream>>>(stats, (float)E, C0.g, C0.b, scaleb, shiftb, 32);
        zero(acc, (size_t)Nn * 32);
        conv5<1><<<dim3((E + 255) / 256), dim3(256), 0, stream>>>(
            A, esrc, edst, C0.tw, C0.tb, C0.pw, C0.pb, nullptr, scaleb, shiftb, acc, E);
        node_finalize<<<dim3((Nn * 32 + 255) / 256), dim3(256), 0, stream>>>(acc, deg, B, Nn, 32, 5);

        run_wmma_conv(convs[1], B, C);
        int tn = Nn * 32;
        residual_k<<<dim3((tn + 255) / 256), dim3(256), 0, stream>>>(SC, C, A, tn);
    }

    // ---- blocks 1..5 (all WMMA) ----
    for (int b = 1; b < 6; ++b) {
        const float* scptr = A;
        if (hasSC[b]) { run_wmma_sc(scw[b], scg[b], scb[b], cins[b], couts[b], A, SC); scptr = SC; }
        run_wmma_conv(convs[2 * b],     A, B);
        run_wmma_conv(convs[2 * b + 1], B, C);
        int tn = Nn * couts[b];
        residual_k<<<dim3((tn + 255) / 256), dim3(256), 0, stream>>>(scptr, C, A, tn);
    }

    // ---- graph mean pool -> [G,128] ----
    zero(pool, (size_t)G * 128);
    scatter_pool<<<dim3((Nn * 128 + 255) / 256), dim3(256), 0, stream>>>(A, ngraph, pool, Nn, 128, 7);
    pool_div<<<dim3((G * 128 + 255) / 256), dim3(256), 0, stream>>>(pool, npg, G * 128, 7);

    // ---- MLP head 128->256->512->1024 ----
    linear_k<true><<<dim3((G * 256 + 255) / 256), dim3(256), 0, stream>>>(pool, hw[0], hb[0], h1, G, 128, 256);
    linear_k<true><<<dim3((G * 512 + 255) / 256), dim3(256), 0, stream>>>(h1, hw[1], hb[1], h2, G, 256, 512);
    linear_k<false><<<dim3((G * 1024 + 255) / 256), dim3(256), 0, stream>>>(h2, hw[2], hb[2], (float*)d_out, G, 512, 1024);

#undef EW_EDGE
#undef EW_NODE
}
